// AttentionLayer_27754078667426
// MI455X (gfx1250) — matrix operand order repaired and checked by reading, not compiled, not run on hardware
//
#include <hip/hip_runtime.h>
#include <hip/hip_bf16.h>

// ---------------------------------------------------------------------------
// CDNA5 (gfx1250) fused attention layer, bf16 WMMA (v_wmma_f32_16x16x32_bf16).
// Big fused GEMM stages 128 V-rows x 1024 K in ~262KB LDS (read V from HBM
// once), B fragments come from a pre-transposed bf16 Wv (2 x b128 per frag).
// ---------------------------------------------------------------------------

typedef __attribute__((ext_vector_type(16))) __bf16 v16bf;
typedef __attribute__((ext_vector_type(8)))  __bf16 v8bf;
typedef __attribute__((ext_vector_type(8)))  float  v8f;

#define D_DIM 1024
#define H_DIM 1024
#define B_DIM 128
#define R_DIM 512
#define NREG  513            // R + 1

__device__ __forceinline__ __bf16 f2bf(float x) {
    union { float f; unsigned u; } v; v.f = x;
    unsigned r = (v.u + 0x7FFFu + ((v.u >> 16) & 1u)) >> 16;   // RNE
    union { unsigned short s; __bf16 b; } o; o.s = (unsigned short)r;
    return o.b;
}

// Two aligned 16B packets -> one 16-element fragment.
template <typename P>
__device__ __forceinline__ v16bf frag16(P lo_p, P hi_p) {
    v8bf lo = *(const v8bf*)lo_p;
    v8bf hi = *(const v8bf*)hi_p;
    return __builtin_shufflevector(lo, hi, 0, 1, 2, 3, 4, 5, 6, 7,
                                           8, 9, 10, 11, 12, 13, 14, 15);
}

// ---------------------------------------------------------------------------
// One-shot: WT[n][k] = bf16(W[k][n])   (1024x1024)
// ---------------------------------------------------------------------------
__global__ __launch_bounds__(256) void transpose_to_bf16(
    const float* __restrict__ W, __bf16* __restrict__ WT)
{
    __shared__ float tile[32][33];
    int bx = blockIdx.x * 32, by = blockIdx.y * 32;
    int tx = threadIdx.x & 31, ty = threadIdx.x >> 5;
    for (int i = ty; i < 32; i += 8)
        tile[i][tx] = W[(size_t)(by + i) * H_DIM + bx + tx];
    __syncthreads();
    for (int i = ty; i < 32; i += 8)
        WT[(size_t)(bx + i) * D_DIM + by + tx] = f2bf(tile[tx][i]);
}

// ---------------------------------------------------------------------------
// Small GEMM: out[128,1024] = ACT((A (+A2)) @ W + bias).
// One wave per 16x64 slab (4 n-tiles share the A fragment).
// ACT: 0 none, 1 relu, 2 tanh.  ADD2/ACT are compile-time (no runtime branch).
// ---------------------------------------------------------------------------
template <bool ADD2, int ACT>
__global__ __launch_bounds__(32) void wmma_gemm_128(
    const float* __restrict__ A, const float* __restrict__ A2,
    const float* __restrict__ W, const float* __restrict__ bias,
    float* __restrict__ out)
{
    const int ng   = blockIdx.x;              // 0..15 -> n-tiles ng*4..ng*4+3
    const int mt   = blockIdx.y;              // 0..7
    const int lane = threadIdx.x & 31;
    const int row  = mt * 16 + (lane & 15);
    const int kbA  = (lane < 16) ? 0 : 8;
    const int kbB  = (lane < 16) ? 0 : 8;
    const int colb = lane & 15;
    const int off4[4] = {0, 1, 4, 5};         // float4 offsets of the 2 K-runs

    v8f acc[4] = {};
    for (int k0 = 0; k0 < D_DIM; k0 += 32) {
        // A fragment: two contiguous 8-float runs -> 4 float4 loads (+A2).
        const float4* p4 = (const float4*)(A + (size_t)row * D_DIM + k0 + kbA);
        v16bf a;
#pragma unroll
        for (int c = 0; c < 4; ++c) {
            float4 f = p4[off4[c]];
            if (ADD2) {
                const float4* q4 =
                    (const float4*)(A2 + (size_t)row * D_DIM + k0 + kbA);
                float4 g = q4[off4[c]];
                f.x += g.x; f.y += g.y; f.z += g.z; f.w += g.w;
            }
            a[c * 4 + 0] = f2bf(f.x); a[c * 4 + 1] = f2bf(f.y);
            a[c * 4 + 2] = f2bf(f.z); a[c * 4 + 3] = f2bf(f.w);
        }
#pragma unroll
        for (int t = 0; t < 4; ++t) {
            int col = (ng * 4 + t) * 16 + colb;
            v16bf bf;
#pragma unroll
            for (int i = 0; i < 16; ++i)
                bf[i] = f2bf(W[(size_t)(k0 + kbB + i + (i & 8)) * H_DIM + col]);
            acc[t] = __builtin_amdgcn_wmma_f32_16x16x32_bf16(
                false, a, false, bf, (short)0, acc[t], false, false);
        }
    }
#pragma unroll
    for (int t = 0; t < 4; ++t) {
        int   col = (ng * 4 + t) * 16 + colb;
        float bsn = bias[col];
#pragma unroll
        for (int j = 0; j < 8; ++j) {
            int   m = mt * 16 + j + ((lane < 16) ? 0 : 8);
            float x = acc[t][j] + bsn;
            if (ACT == 1)      x = fmaxf(x, 0.f);
            else if (ACT == 2) x = tanhf(x);
            out[(size_t)m * H_DIM + col] = x;
        }
    }
}

// ---------------------------------------------------------------------------
// scores[b,r] = tanh(relu(V[b,r]@Wv + bv) + h_att[b]) . Wal + bal
// Block = 256 threads (8 waves), 128 flat V rows staged once in LDS as bf16.
// Wave w: m-tiles (w>>2)*4 .. +3, n-tiles (w&3)*16 .. +15.
// B frag = 2 x global b128 from WvT; A frag = 2 x ds b128 (LDA-padded, bank
// conflict free). Per-row scores combined across waves with ds_add_f32.
// ---------------------------------------------------------------------------
#define LDA 1048   // bf16 row stride: 524 dwords -> 16 rows on 16 distinct banks

__global__ __launch_bounds__(256) void scores_kernel(
    const float* __restrict__ V,   const __bf16* __restrict__ WvT,
    const float* __restrict__ bv,  const float* __restrict__ h_att,
    const float* __restrict__ Wal, const float* __restrict__ bal,
    float* __restrict__ scores)
{
    __shared__ __bf16 As[128 * LDA];          // ~262 KB
    __shared__ float  sred[128];

    const int tid  = threadIdx.x;
    const int lane = tid & 31;
    const int wave = tid >> 5;
    const int m0   = blockIdx.x * 128;        // 512 blocks over B*R rows
    const int b    = m0 >> 9;
    const int r0   = m0 & (R_DIM - 1);

    // Stage A (128 x 1024) as bf16, float4 in -> v8bf (16B) LDS stores.
    const float4* Vf4 = (const float4*)(V + (size_t)m0 * D_DIM);
    for (int i = tid; i < 128 * D_DIM / 8; i += 256) {
        float4 f0 = Vf4[i * 2], f1 = Vf4[i * 2 + 1];
        int e = i * 8, row = e >> 10, k = e & (D_DIM - 1);
        v8bf pk;
        pk[0] = f2bf(f0.x); pk[1] = f2bf(f0.y); pk[2] = f2bf(f0.z); pk[3] = f2bf(f0.w);
        pk[4] = f2bf(f1.x); pk[5] = f2bf(f1.y); pk[6] = f2bf(f1.z); pk[7] = f2bf(f1.w);
        *(v8bf*)&As[row * LDA + k] = pk;
    }
    if (tid < 128) sred[tid] = bal[0];
    __syncthreads();

    const int g    = wave >> 2;               // m-tile group (0/1)
    const int nset = wave & 3;                // n-tile set   (0..3)
    const int kbA  = (lane < 16) ? 0 : 8;
    const int kbB  = (lane < 16) ? 0 : 8;
    const int rlo  = lane & 15;

    float scorep[4][8];
#pragma unroll
    for (int mi = 0; mi < 4; ++mi)
#pragma unroll
        for (int j = 0; j < 8; ++j) scorep[mi][j] = 0.f;

    for (int t = 0; t < 16; ++t) {
        const int     nt = nset * 16 + t;
        const int     n0 = nt * 16;
        const __bf16* bp = WvT + (size_t)(n0 + rlo) * D_DIM + kbB;

        v8f acc[4] = {};
        for (int k0 = 0; k0 < D_DIM; k0 += 32) {
            v16bf bf = frag16(bp + k0, bp + k0 + 16);
#pragma unroll
            for (int mi = 0; mi < 4; ++mi) {
                const __bf16* ap =
                    &As[((g * 4 + mi) * 16 + rlo) * LDA + k0 + kbA];
                v16bf a = frag16(ap, ap + 16);
                acc[mi] = __builtin_amdgcn_wmma_f32_16x16x32_bf16(
                    false, a, false, bf, (short)0, acc[mi], false, false);
            }
        }
        const int   n    = n0 + rlo;
        const float bvn  = bv[n];
        const float han  = h_att[(size_t)b * H_DIM + n];
        const float waln = Wal[n];
#pragma unroll
        for (int mi = 0; mi < 4; ++mi)
#pragma unroll
            for (int j = 0; j < 8; ++j) {
                float x = fmaxf(acc[mi][j] + bvn, 0.f) + han;
                scorep[mi][j] += tanhf(x) * waln;
            }
    }

    // Lane-reduce inside each 16-lane half of the C layout, combine via LDS.
#pragma unroll
    for (int mi = 0; mi < 4; ++mi)
#pragma unroll
        for (int j = 0; j < 8; ++j) {
            float s = scorep[mi][j];
            s += __shfl_xor(s, 1, 32);
            s += __shfl_xor(s, 2, 32);
            s += __shfl_xor(s, 4, 32);
            s += __shfl_xor(s, 8, 32);
            scorep[mi][j] = s;
        }
    if ((lane & 15) == 0) {
        int half = (lane >> 4) << 3;
#pragma unroll
        for (int mi = 0; mi < 4; ++mi)
#pragma unroll
            for (int j = 0; j < 8; ++j)
                atomicAdd(&sred[(g * 4 + mi) * 16 + half + j], scorep[mi][j]);
    }
    __syncthreads();
    if (tid < 128) scores[(size_t)b * NREG + r0 + tid] = sred[tid];
}

// ---------------------------------------------------------------------------
// scores[b, R] = tanh(s_att[b] + h_att[b]) . Wal + bal
// ---------------------------------------------------------------------------
__global__ __launch_bounds__(256) void score_s_kernel(
    const float* __restrict__ s_att, const float* __restrict__ h_att,
    const float* __restrict__ Wal,   const float* __restrict__ bal,
    float* __restrict__ scores)
{
    __shared__ float sred[256];
    int b = blockIdx.x, tid = threadIdx.x;
    float p = 0.f;
    for (int n = tid; n < H_DIM; n += 256)
        p += tanhf(s_att[(size_t)b * H_DIM + n] + h_att[(size_t)b * H_DIM + n]) * Wal[n];
    sred[tid] = p; __syncthreads();
    for (int s = 128; s > 0; s >>= 1) {
        if (tid < s) sred[tid] += sred[tid + s];
        __syncthreads();
    }
    if (tid == 0) scores[(size_t)b * NREG + R_DIM] = sred[0] + bal[0];
}

// ---------------------------------------------------------------------------
// softmax over 513 scores, then context[b] = sum_r alpha_r * regions[b,r]
// ---------------------------------------------------------------------------
__global__ __launch_bounds__(256) void softmax_context_kernel(
    const float* __restrict__ scores, const float* __restrict__ V,
    const float* __restrict__ s_proj, float* __restrict__ context)
{
    __shared__ float alpha[NREG];
    __shared__ float sred[256];
    int b = blockIdx.x, tid = threadIdx.x;

    float m = -3.4e38f;
    for (int i = tid; i < NREG; i += 256) {
        float s = scores[(size_t)b * NREG + i];
        alpha[i] = s;
        m = fmaxf(m, s);
    }
    sred[tid] = m; __syncthreads();
    for (int s = 128; s > 0; s >>= 1) {
        if (tid < s) sred[tid] = fmaxf(sred[tid], sred[tid + s]);
        __syncthreads();
    }
    m = sred[0]; __syncthreads();

    float sum = 0.f;
    for (int i = tid; i < NREG; i += 256) {
        float e = __expf(alpha[i] - m);
        alpha[i] = e;
        sum += e;
    }
    sred[tid] = sum; __syncthreads();
    for (int s = 128; s > 0; s >>= 1) {
        if (tid < s) sred[tid] += sred[tid + s];
        __syncthreads();
    }
    float inv = 1.f / sred[0]; __syncthreads();

    const float4* Vb = (const float4*)(V + (size_t)b * R_DIM * D_DIM);
    float cx = 0.f, cy = 0.f, cz = 0.f, cw = 0.f;
    for (int r = 0; r < R_DIM; ++r) {
        float  a = alpha[r] * inv;
        float4 v = Vb[r * (D_DIM / 4) + tid];
        cx += a * v.x; cy += a * v.y; cz += a * v.z; cw += a * v.w;
    }
    float  as = alpha[R_DIM] * inv;
    float4 sp = ((const float4*)(s_proj + (size_t)b * D_DIM))[tid];
    cx += as * sp.x; cy += as * sp.y; cz += as * sp.z; cw += as * sp.w;
    float4 o = {cx, cy, cz, cw};
    ((float4*)(context + (size_t)b * D_DIM))[tid] = o;
}

// ---------------------------------------------------------------------------
extern "C" void kernel_launch(void* const* d_in, const int* in_sizes, int n_in,
                              void* d_out, int out_size, void* d_ws, size_t ws_size,
                              hipStream_t stream) {
    const float* V   = (const float*)d_in[0];
    const float* s_t = (const float*)d_in[1];
    const float* h_t = (const float*)d_in[2];
    const float* Wv  = (const float*)d_in[3];
    const float* bv  = (const float*)d_in[4];
    const float* Wsp = (const float*)d_in[5];
    const float* bsp = (const float*)d_in[6];
    const float* Wsa = (const float*)d_in[7];
    const float* bsa = (const float*)d_in[8];
    const float* Whp = (const float*)d_in[9];
    const float* bhp = (const float*)d_in[10];
    const float* Wha = (const float*)d_in[11];
    const float* bha = (const float*)d_in[12];
    const float* Wal = (const float*)d_in[13];
    const float* bal = (const float*)d_in[14];
    const float* Wcp = (const float*)d_in[15];
    const float* bcp = (const float*)d_in[16];

    float*   ws      = (float*)d_ws;
    float*   s_proj  = ws;                       // 128*1024
    float*   h_proj  = s_proj + B_DIM * D_DIM;
    float*   s_att   = h_proj + B_DIM * D_DIM;
    float*   h_att   = s_att  + B_DIM * H_DIM;
    float*   scores  = h_att  + B_DIM * H_DIM;   // 128*513
    float*   context = scores + B_DIM * NREG;
    __bf16*  WvT     = (__bf16*)(context + B_DIM * D_DIM);  // 1024*1024 bf16

    float* z_t = (float*)d_out;

    // Wv -> bf16 transpose (B frags become contiguous-K b128 loads).
    transpose_to_bf16<<<dim3(32, 32), 256, 0, stream>>>(Wv, WvT);

    dim3 gGemm(16, 8);
    wmma_gemm_128<false, 1><<<gGemm, 32, 0, stream>>>(s_t, nullptr, Wsp, bsp, s_proj);
    wmma_gemm_128<false, 2><<<gGemm, 32, 0, stream>>>(h_t, nullptr, Whp, bhp, h_proj);
    wmma_gemm_128<false, 0><<<gGemm, 32, 0, stream>>>(s_proj, nullptr, Wsa, bsa, s_att);
    wmma_gemm_128<false, 0><<<gGemm, 32, 0, stream>>>(h_proj, nullptr, Wha, bha, h_att);

    scores_kernel<<<(B_DIM * R_DIM) / 128, 256, 0, stream>>>(
        V, WvT, bv, h_att, Wal, bal, scores);
    score_s_kernel<<<B_DIM, 256, 0, stream>>>(s_att, h_att, Wal, bal, scores);

    softmax_context_kernel<<<B_DIM, 256, 0, stream>>>(scores, V, s_proj, context);

    wmma_gemm_128<true, 2><<<gGemm, 32, 0, stream>>>(context, h_proj, Wcp, bcp, z_t);
}
